// AttentionMC_87600152969886
// MI455X (gfx1250) — matrix operand order repaired and checked by reading, not compiled, not run on hardware
//
#include <hip/hip_runtime.h>
#include <hip/hip_bf16.h>

// ---------------------------------------------------------------------------
// CDNA5 / gfx1250 multi-head causal attention forward (bf16 WMMA, fp32 accum)
// B=2, S=2048, D=1024, H=16, hd=64
// Pipeline:
//   1) one-shot fp32 -> bf16 conversion of x and all weights
//   2) TDM-staged bf16 GEMMs (tensor_load_to_lds, double buffered)
//   3) RoPE (fp32 math, bf16 in place)
//   4) flash-style causal attention, all matmuls on v_wmma_f32_16x16x32_bf16
//   5) TDM-staged output projection -> f32
// ---------------------------------------------------------------------------

typedef __bf16 bf16_t;
typedef __attribute__((ext_vector_type(16))) bf16_t v16bf;
typedef __attribute__((ext_vector_type(8)))  bf16_t v8bf;
typedef __attribute__((ext_vector_type(8)))  float  v8f;
typedef __attribute__((ext_vector_type(4)))  unsigned int v4u;
typedef __attribute__((ext_vector_type(8)))  unsigned int v8u;

#define EMBED_DIM 1024
#define NUM_HEADS 16
#define HEAD_DIM  64
#define SEQ_LEN   2048
#define BATCH     2

static __device__ __forceinline__ v16bf join16(v8bf lo, v8bf hi) {
    return __builtin_shufflevector(lo, hi, 0,1,2,3,4,5,6,7,8,9,10,11,12,13,14,15);
}
static __device__ __forceinline__ v8bf ld8(const bf16_t* p) {
    return *(const v8bf*)p;
}
static __device__ __forceinline__ v8f wmma_bf16(v16bf a, v16bf b, v8f c) {
    return __builtin_amdgcn_wmma_f32_16x16x32_bf16(
        false, a, false, b, (short)0, c, false, false);
}

// ---- Tensor Data Mover descriptor helpers (ISA cdna5 §8.3/§8.4) ----
static __device__ __forceinline__ v4u tdm_g0(unsigned lds_addr, unsigned long long gaddr) {
    v4u g;
    g[0] = 1u;                                   // count=1, user mode, no gather
    g[1] = lds_addr;                             // LDS byte address
    g[2] = (unsigned)gaddr;                      // global_addr[31:0]
    g[3] = (unsigned)(gaddr >> 32) | 0x80000000u; // global_addr[56:32] | type=2
    return g;
}
static __device__ __forceinline__ v8u tdm_g1(unsigned dim0, unsigned dim1,
                                             unsigned tile0, unsigned tile1,
                                             unsigned stride0) {
    v8u g;
    g[0] = 0x10000u;                              // data_size=1 -> 2-byte elems
    g[1] = (dim0 & 0xffffu) << 16;                // tensor_dim0[15:0]
    g[2] = (dim0 >> 16) | ((dim1 & 0xffffu) << 16);
    g[3] = (dim1 >> 16) | (tile0 << 16);          // tile_dim0
    g[4] = tile1;                                 // tile_dim1 (tile_dim2=0)
    g[5] = stride0;                               // tensor_dim0_stride[31:0]
    g[6] = 0;                                     // stride hi / dim1_stride lo
    g[7] = 0;
    return g;
}
static __device__ __forceinline__ void tdm_load(v4u g0, v8u g1) {
    // 2-D tensor: 2 SGPR-group form (VADDR2/VADDR3 = NULL)
    asm volatile("tensor_load_to_lds %0, %1" :: "s"(g0), "s"(g1) : "memory");
}
static __device__ __forceinline__ unsigned lds_off(const void* p) {
    return (unsigned)(unsigned long long)p;       // LDS aperture: low 32 bits
}

// ---------------------------------------------------------------------------
// One-shot fp32 -> bf16 conversion (4 elems / thread)
// ---------------------------------------------------------------------------
__global__ __launch_bounds__(256)
void cvt_f32_bf16(const float* __restrict__ in, bf16_t* __restrict__ out, int n4) {
    int i = blockIdx.x * blockDim.x + threadIdx.x;
    if (i >= n4) return;
    float4 f = *(const float4*)(in + (size_t)i * 4);
    bf16_t* d = out + (size_t)i * 4;
    d[0] = (bf16_t)f.x; d[1] = (bf16_t)f.y; d[2] = (bf16_t)f.z; d[3] = (bf16_t)f.w;
}

// ---------------------------------------------------------------------------
// GEMM: C[M,N] = A[M,K] (bf16) @ W[N,K]^T (bf16), fp32 accumulate.
// 256 threads (8 waves), 128x128 tile, K-step 32.
// LDS tiles staged by the Tensor Data Mover, double buffered:
//   wave 0 issues next A/B tile DMA, waits TENSORcnt<=2 for current pair,
//   barrier publishes; inner loop is pure ds_load_b128 + v_wmma.
// mode 0: f32 row-major [M,N]
// mode 1: bf16 scattered to [B,H,S,hd]   (q / k layout)
// mode 2: bf16 scattered to [B,H,hd,S]   (v transposed, PV B-operand)
// ---------------------------------------------------------------------------
__global__ __launch_bounds__(256)
void gemm_nt_bf16(const bf16_t* __restrict__ A, const bf16_t* __restrict__ W,
                  void* __restrict__ out, int M, int N, int K, int mode) {
    __shared__ bf16_t lA[2][128 * 32];
    __shared__ bf16_t lB[2][128 * 32];

    const int tid  = threadIdx.x;
    const int lane = tid & 31;
    const int wid  = tid >> 5;
    const int wm   = wid & 3;        // wave row strip: 32 rows
    const int wn   = wid >> 2;       // wave col strip: 64 cols
    const int gl   = lane >> 4;
    const int nloc = lane & 15;
    const int row0 = blockIdx.y * 128;
    const int col0 = blockIdx.x * 128;

    v8f acc[2][4];
    #pragma unroll
    for (int mt = 0; mt < 2; ++mt)
        #pragma unroll
        for (int nt = 0; nt < 4; ++nt)
            acc[mt][nt] = (v8f){0.f,0.f,0.f,0.f,0.f,0.f,0.f,0.f};

    const int steps = K >> 5;                      // K/32
    const unsigned long long aBase = (unsigned long long)A + ((size_t)row0 * K) * 2ull;
    const unsigned long long wBase = (unsigned long long)W + ((size_t)col0 * K) * 2ull;

    if (wid == 0) {                                // prologue: tile 0
        tdm_load(tdm_g0(lds_off(&lA[0][0]), aBase), tdm_g1(K, M, 32, 128, K));
        tdm_load(tdm_g0(lds_off(&lB[0][0]), wBase), tdm_g1(K, N, 32, 128, K));
    }

    for (int i = 0; i < steps; ++i) {
        const int b = i & 1;
        if (wid == 0) {
            if (i + 1 < steps) {                   // prefetch tile i+1 via TDM
                unsigned long long off = ((unsigned long long)(i + 1) * 32ull) * 2ull;
                tdm_load(tdm_g0(lds_off(&lA[b ^ 1][0]), aBase + off), tdm_g1(K, M, 32, 128, K));
                tdm_load(tdm_g0(lds_off(&lB[b ^ 1][0]), wBase + off), tdm_g1(K, N, 32, 128, K));
                __builtin_amdgcn_s_wait_tensorcnt(2);  // pair i done, pair i+1 in flight
            } else {
                __builtin_amdgcn_s_wait_tensorcnt(0);
            }
        }
        __syncthreads();                           // tile i visible to all waves

        // ---- A fragments (ISA 16-bit A layout: chunks at 8g and 16+8g) ----
        v16bf afrag[2];
        #pragma unroll
        for (int mt = 0; mt < 2; ++mt) {
            const bf16_t* base = &lA[b][(wm * 32 + mt * 16 + nloc) * 32];
            afrag[mt] = join16(ld8(base + 8 * gl), ld8(base + 16 + 8 * gl));
        }
        // ---- B fragments (ISA 16-bit layout: chunks at 8g and 16+8g of column n) ----
        v16bf bfrag[4];
        #pragma unroll
        for (int nt = 0; nt < 4; ++nt) {
            const bf16_t* base = &lB[b][(wn * 64 + nt * 16 + nloc) * 32];
            bfrag[nt] = join16(ld8(base + 8 * gl), ld8(base + 16 + 8 * gl));
        }
        #pragma unroll
        for (int mt = 0; mt < 2; ++mt)
            #pragma unroll
            for (int nt = 0; nt < 4; ++nt)
                acc[mt][nt] = wmma_bf16(afrag[mt], bfrag[nt], acc[mt][nt]);

        __syncthreads();                           // reads done before TDM reuses buffer
    }

    // ---- epilogue: C layout lane holds (N=nloc, M=r+8*gl) ----
    #pragma unroll
    for (int mt = 0; mt < 2; ++mt) {
        #pragma unroll
        for (int nt = 0; nt < 4; ++nt) {
            #pragma unroll
            for (int r = 0; r < 8; ++r) {
                int grow = row0 + wm * 32 + mt * 16 + gl * 8 + r;
                int gcol = col0 + wn * 64 + nt * 16 + nloc;
                float val = acc[mt][nt][r];
                if (mode == 0) {
                    ((float*)out)[(size_t)grow * N + gcol] = val;
                } else {
                    int bb = grow / SEQ_LEN, s = grow - bb * SEQ_LEN;
                    int h = gcol >> 6, d = gcol & 63;
                    if (mode == 1) {
                        ((bf16_t*)out)[(((size_t)bb * NUM_HEADS + h) * SEQ_LEN + s) * HEAD_DIM + d] = (bf16_t)val;
                    } else {
                        ((bf16_t*)out)[(((size_t)bb * NUM_HEADS + h) * HEAD_DIM + d) * SEQ_LEN + s] = (bf16_t)val;
                    }
                }
            }
        }
    }
}

// ---------------------------------------------------------------------------
// RoPE in place on bf16 [B,H,S,hd]; one thread per (row, freq) pair.
// ---------------------------------------------------------------------------
__global__ __launch_bounds__(256)
void rope_kernel(bf16_t* __restrict__ t) {
    int idx = blockIdx.x * blockDim.x + threadIdx.x;  // B*H*S*32 threads
    int f   = idx & 31;
    long row = idx >> 5;
    int  s  = (int)(row % SEQ_LEN);
    float inv = __powf(10000.0f, -(float)f * (1.0f / 32.0f));
    float ang = (float)s * inv;
    float sn, cs;
    __sincosf(ang, &sn, &cs);
    bf16_t* p = t + row * HEAD_DIM;
    float t1 = (float)p[f];
    float t2 = (float)p[32 + f];
    p[f]      = (bf16_t)(t1 * cs - t2 * sn);
    p[32 + f] = (bf16_t)(t1 * sn + t2 * cs);
}

// ---------------------------------------------------------------------------
// Flash-style causal attention. One wave per 16-query tile per (b,h).
// q,k: bf16 [B,H,S,64]; vt: bf16 [B,H,64,S]; out: bf16 [B,S,1024].
// ---------------------------------------------------------------------------
__global__ __launch_bounds__(32)
void attn_kernel(const bf16_t* __restrict__ q, const bf16_t* __restrict__ k,
                 const bf16_t* __restrict__ vt, bf16_t* __restrict__ o) {
    __shared__ bf16_t P[16 * 32];

    const int lane = threadIdx.x;
    const int gl   = lane >> 4;
    const int nloc = lane & 15;

    const int tilesPerSeq = SEQ_LEN / 16;
    const int bh = blockIdx.x / tilesPerSeq;
    const int qs = (blockIdx.x % tilesPerSeq) * 16;

    const bf16_t* qbase = q  + (size_t)bh * SEQ_LEN * HEAD_DIM;
    const bf16_t* kbase = k  + (size_t)bh * SEQ_LEN * HEAD_DIM;
    const bf16_t* vbase = vt + (size_t)bh * HEAD_DIM * SEQ_LEN;

    v16bf qf[2];
    #pragma unroll
    for (int ks2 = 0; ks2 < 2; ++ks2) {
        const bf16_t* base = qbase + (size_t)(qs + nloc) * HEAD_DIM + ks2 * 32;
        qf[ks2] = join16(ld8(base + 8 * gl), ld8(base + 16 + 8 * gl));
    }

    float mstate[8], lstate[8];
    v8f acc[4];
    #pragma unroll
    for (int r = 0; r < 8; ++r) { mstate[r] = -1e30f; lstate[r] = 0.f; }
    #pragma unroll
    for (int nt = 0; nt < 4; ++nt) acc[nt] = (v8f){0.f,0.f,0.f,0.f,0.f,0.f,0.f,0.f};

    const int jmax = (qs + 15) >> 5;
    for (int j = 0; j <= jmax; ++j) {
        const int ks = j << 5;

        v8f sc[2];
        sc[0] = (v8f){0.f,0.f,0.f,0.f,0.f,0.f,0.f,0.f};
        sc[1] = (v8f){0.f,0.f,0.f,0.f,0.f,0.f,0.f,0.f};
        #pragma unroll
        for (int t = 0; t < 2; ++t) {
            #pragma unroll
            for (int ks2 = 0; ks2 < 2; ++ks2) {
                const bf16_t* base =
                    kbase + (size_t)(ks + t * 16 + nloc) * HEAD_DIM + ks2 * 32;
                v16bf bf = join16(ld8(base + 8 * gl), ld8(base + 16 + 8 * gl));
                sc[t] = wmma_bf16(qf[ks2], bf, sc[t]);
            }
        }
        if (j < jmax)
            __builtin_prefetch(kbase + (size_t)(ks + 32 + nloc) * HEAD_DIM, 0, 1);

        #pragma unroll
        for (int r = 0; r < 8; ++r) {
            const int qrow = qs + gl * 8 + r;
            float s0 = sc[0][r] * 0.125f;
            float s1 = sc[1][r] * 0.125f;
            if (ks + nloc      > qrow) s0 = -1e30f;
            if (ks + 16 + nloc > qrow) s1 = -1e30f;
            float mx = fmaxf(s0, s1);
            #pragma unroll
            for (int off = 8; off; off >>= 1)
                mx = fmaxf(mx, __shfl_xor(mx, off, 16));
            float mnew = fmaxf(mstate[r], mx);
            float corr = __expf(mstate[r] - mnew);
            float p0 = __expf(s0 - mnew);
            float p1 = __expf(s1 - mnew);
            float rs = p0 + p1;
            #pragma unroll
            for (int off = 8; off; off >>= 1)
                rs += __shfl_xor(rs, off, 16);
            lstate[r] = lstate[r] * corr + rs;
            mstate[r] = mnew;
            #pragma unroll
            for (int nt = 0; nt < 4; ++nt) acc[nt][r] *= corr;
            P[(gl * 8 + r) * 32 + nloc]      = (bf16_t)p0;
            P[(gl * 8 + r) * 32 + 16 + nloc] = (bf16_t)p1;
        }
        __syncthreads();

        const bf16_t* pb = &P[nloc * 32];
        v16bf pf = join16(ld8(pb + 8 * gl), ld8(pb + 16 + 8 * gl));

        #pragma unroll
        for (int nt = 0; nt < 4; ++nt) {
            const bf16_t* base =
                vbase + (size_t)(nt * 16 + nloc) * SEQ_LEN + ks;
            v16bf vf = join16(ld8(base + 8 * gl), ld8(base + 16 + 8 * gl));
            acc[nt] = wmma_bf16(pf, vf, acc[nt]);
        }
        __syncthreads();
    }

    const int b = bh >> 4, h = bh & 15;
    #pragma unroll
    for (int nt = 0; nt < 4; ++nt) {
        #pragma unroll
        for (int r = 0; r < 8; ++r) {
            int orow = qs + gl * 8 + r;
            int d    = nt * 16 + nloc;
            o[((size_t)b * SEQ_LEN + orow) * EMBED_DIM + h * HEAD_DIM + d] =
                (bf16_t)(acc[nt][r] / lstate[r]);
        }
    }
}

// ---------------------------------------------------------------------------
// Host-side orchestration
// ---------------------------------------------------------------------------
extern "C" void kernel_launch(void* const* d_in, const int* in_sizes, int n_in,
                              void* d_out, int out_size, void* d_ws, size_t ws_size,
                              hipStream_t stream) {
    const float* x  = (const float*)d_in[0];
    const float* Wq = (const float*)d_in[1];
    const float* Wk = (const float*)d_in[2];
    const float* Wv = (const float*)d_in[3];
    const float* Wo = (const float*)d_in[4];

    const int M = BATCH * SEQ_LEN;        // 4096
    const int N = EMBED_DIM;              // 1024
    const int K = EMBED_DIM;              // 1024

    const size_t xElems = (size_t)M * K;                                   // 4.19M
    const size_t wElems = (size_t)N * K;                                   // 1.05M
    const size_t qkvElems = (size_t)BATCH * NUM_HEADS * SEQ_LEN * HEAD_DIM; // 4.19M

    char* ws = (char*)d_ws;
    bf16_t* xb   = (bf16_t*)(ws);                                  size_t off = xElems * 2;
    bf16_t* wqb  = (bf16_t*)(ws + off);  off += wElems * 2;
    bf16_t* wkb  = (bf16_t*)(ws + off);  off += wElems * 2;
    bf16_t* wvb  = (bf16_t*)(ws + off);  off += wElems * 2;
    bf16_t* wob  = (bf16_t*)(ws + off);  off += wElems * 2;
    bf16_t* qb   = (bf16_t*)(ws + off);  off += qkvElems * 2;
    bf16_t* kb   = (bf16_t*)(ws + off);  off += qkvElems * 2;
    bf16_t* vtb  = (bf16_t*)(ws + off);  off += qkvElems * 2;
    bf16_t* atto = (bf16_t*)(ws + off);                            // [B,S,D] bf16

    // 1) one-shot conversions to bf16
    cvt_f32_bf16<<<(int)(xElems / 4 / 256), 256, 0, stream>>>(x,  xb,  (int)(xElems / 4));
    cvt_f32_bf16<<<(int)(wElems / 4 / 256), 256, 0, stream>>>(Wq, wqb, (int)(wElems / 4));
    cvt_f32_bf16<<<(int)(wElems / 4 / 256), 256, 0, stream>>>(Wk, wkb, (int)(wElems / 4));
    cvt_f32_bf16<<<(int)(wElems / 4 / 256), 256, 0, stream>>>(Wv, wvb, (int)(wElems / 4));
    cvt_f32_bf16<<<(int)(wElems / 4 / 256), 256, 0, stream>>>(Wo, wob, (int)(wElems / 4));

    dim3 gemmGrid(N / 128, M / 128);      // 8 x 32
    // 2) projections (TDM-staged bf16 GEMMs)
    gemm_nt_bf16<<<gemmGrid, 256, 0, stream>>>(xb, wqb, qb,  M, N, K, 1);
    gemm_nt_bf16<<<gemmGrid, 256, 0, stream>>>(xb, wkb, kb,  M, N, K, 1);
    gemm_nt_bf16<<<gemmGrid, 256, 0, stream>>>(xb, wvb, vtb, M, N, K, 2);

    // 3) RoPE on q and k
    const int ropeThreads = BATCH * NUM_HEADS * SEQ_LEN * 32;  // 2.1M
    rope_kernel<<<ropeThreads / 256, 256, 0, stream>>>(qb);
    rope_kernel<<<ropeThreads / 256, 256, 0, stream>>>(kb);

    // 4) causal flash attention
    const int attnBlocks = BATCH * NUM_HEADS * (SEQ_LEN / 16);  // 4096
    attn_kernel<<<attnBlocks, 32, 0, stream>>>(qb, kb, vtb, atto);

    // 5) output projection -> f32 d_out
    gemm_nt_bf16<<<gemmGrid, 256, 0, stream>>>(atto, wob, d_out, M, N, K, 0);
}
